// HierarchicalMoE_1520418423058
// MI455X (gfx1250) — matrix operand order repaired and checked by reading, not compiled, not run on hardware
//
#include <hip/hip_runtime.h>
#include <math.h>

typedef unsigned short     u16;
typedef unsigned int       u32;
typedef unsigned long long u64;

typedef __bf16 v16bf __attribute__((ext_vector_type(16)));
typedef float  v8f   __attribute__((ext_vector_type(8)));

union FragBF {
  uint4 u[2];
  v16bf v;
};

#define D      512
#define H2     1024
#define NPAIR  64
#define MAXTOK 1024

#define AP     520           // A-tile pitch (bf16): 1040B = 65*16
#define HP     1032          // H-tile pitch (bf16): 2064B = 129*16
#define BPITCH 132           // weight-tile pitch (fp32 dwords): 528B = 33*16
#define BUFELT (64 * BPITCH) // fp32 elements per ping-pong buffer (64 k-rows)

__device__ __forceinline__ u16 f2bf(float f) {
  union { float f; u32 u; } x; x.f = f;
  u32 r = x.u + 0x7FFFu + ((x.u >> 16) & 1u);   // round-to-nearest-even
  return (u16)(r >> 16);
}

// Async-DMA one 64-row x 128-col fp32 tile (row stride ldw) into LDS.
// One GLOBAL_LOAD_ASYNC_TO_LDS_B128 per row (32 lanes x 16B = 512B = full row);
// wave w copies rows w*8 .. w*8+7. Tracked by ASYNCcnt (8 per wave per stage).
__device__ __forceinline__ void stage_async(const float* __restrict__ tile,
                                            int ldw, u32 ldsBase, int w, int l) {
#pragma unroll
  for (int j = 0; j < 8; ++j) {
    int r = w * 8 + j;
    u64 ga = (u64)(const void*)(tile + (size_t)r * ldw + l * 4);
    u32 la = ldsBase + (u32)(r * (BPITCH * 4) + l * 16);
    asm volatile("global_load_async_to_lds_b128 %0, %1, off"
                 :: "v"(la), "v"(ga) : "memory");
  }
}

// ---------------------------------------------------------------------------
// Kernel 0: zero output accumulator + per-(g,e) counters
// ---------------------------------------------------------------------------
__global__ __launch_bounds__(256) void moe_zero(float* __restrict__ out, int n,
                                                int* __restrict__ cnt) {
  int i = blockIdx.x * 256 + threadIdx.x;
  if (i < n) out[i] = 0.f;
  if (i < NPAIR) cnt[i] = 0;
}

// ---------------------------------------------------------------------------
// Kernel 1: two-level routing. One 64-thread block per token.
// ---------------------------------------------------------------------------
__global__ __launch_bounds__(64) void moe_route(
    const float* __restrict__ x, const float* __restrict__ g1,
    const float* __restrict__ g2, int* __restrict__ cnt,
    int* __restrict__ tokIdx, float* __restrict__ tokW) {
  const int t = blockIdx.x;
  const int tid = threadIdx.x;
  __shared__ float xs[D];
  __shared__ float l1[8];
  __shared__ float l2[64];

  for (int i = tid; i < D; i += 64) xs[i] = x[(size_t)t * D + i];
  __syncthreads();

  {  // logits2[g][e] = x . g2[g,:,e]
    int g = tid >> 3, e = tid & 7;
    float a = 0.f;
    for (int k = 0; k < D; ++k) a += xs[k] * g2[((size_t)g * D + k) * 8 + e];
    l2[tid] = a;
  }
  if (tid < 8) {  // logits1[g] = x . g1[:,g]
    float a = 0.f;
    for (int k = 0; k < D; ++k) a += xs[k] * g1[(size_t)k * 8 + tid];
    l1[tid] = a;
  }
  __syncthreads();

  if (tid == 0) {
    int bi = 0; float bv = l1[0];
    for (int g = 1; g < 8; ++g) if (l1[g] > bv) { bv = l1[g]; bi = g; }
    int si = (bi == 0) ? 1 : 0; float sv = -3.4e38f;
    for (int g = 0; g < 8; ++g) if (g != bi && l1[g] > sv) { sv = l1[g]; si = g; }
    float e2 = __expf(sv - bv);
    float inv = 1.f / (1.f + e2);
    float wg[2] = { inv, e2 * inv };
    int   gi[2] = { bi, si };
    for (int s = 0; s < 2; ++s) {
      int g = gi[s]; float cw = wg[s];
      const float* r = &l2[g * 8];
      int ei = 0; float ev = r[0];
      for (int e = 1; e < 8; ++e) if (r[e] > ev) { ev = r[e]; ei = e; }
      int fi = (ei == 0) ? 1 : 0; float fv = -3.4e38f;
      for (int e = 0; e < 8; ++e) if (e != ei && r[e] > fv) { fv = r[e]; fi = e; }
      float q  = __expf(fv - ev);
      float qi = 1.f / (1.f + q);
      int p0 = g * 8 + ei, p1 = g * 8 + fi;
      int s0 = atomicAdd(&cnt[p0], 1);
      tokIdx[p0 * MAXTOK + s0] = t; tokW[p0 * MAXTOK + s0] = cw * qi;
      int s1 = atomicAdd(&cnt[p1], 1);
      tokIdx[p1 * MAXTOK + s1] = t; tokW[p1 * MAXTOK + s1] = cw * q * qi;
    }
  }
}

// ---------------------------------------------------------------------------
// Kernel 2: sparse grouped expert MLP. bf16 WMMA, f32 accumulate.
// Weights streamed fp32 via async Global->LDS DMA, double-buffered;
// fp32->bf16 conversion happens at fragment-load time only.
// grid = (token_tiles, 64 pairs), block = 256 threads (8 wave32).
// ---------------------------------------------------------------------------
__global__ __launch_bounds__(256) void moe_expert(
    const float* __restrict__ x,
    const float* __restrict__ W1, const float* __restrict__ b1,
    const float* __restrict__ W2, const float* __restrict__ b2,
    const int* __restrict__ cnt, const int* __restrict__ tokIdx,
    const float* __restrict__ tokW, float* __restrict__ out) {
  const int pair  = blockIdx.y;
  const int tile  = blockIdx.x;
  const int count = cnt[pair];
  if (tile * 16 >= count) return;

  const int tid = threadIdx.x;
  const int w   = tid >> 5;       // wave id 0..7
  const int l   = tid & 31;       // lane
  const int ln  = l & 15;
  const int hi  = l >> 4;         // half-wave select

  __shared__ __align__(16) u16   Abf[16 * AP];      // x tile  bf16 [m][k]
  __shared__ __align__(16) u16   Hs [16 * HP];      // h tile  bf16 [m][k]
  __shared__ __align__(16) float Bfp[2][BUFELT];    // W tile  fp32 [k][n], ping-pong
  __shared__ int   sTok[16];
  __shared__ float sWt[16];

  const u32 bBase0 = (u32)(uintptr_t)(&Bfp[0][0]);
  const u32 bBase1 = (u32)(uintptr_t)(&Bfp[1][0]);

  if (tid < 16) {
    int i = tile * 16 + tid;
    bool ok = i < count;
    sTok[tid] = ok ? tokIdx[pair * MAXTOK + i] : 0;
    sWt[tid]  = ok ? tokW[pair * MAXTOK + i]  : 0.f;
  }
  __syncthreads();

  // gather A tile: 16 tokens x 512, fp32 -> bf16 (once per block)
  for (int i = 0; i < 32; ++i) {
    int m = i >> 1;
    int c = ((i & 1) << 8) + tid;
    Abf[m * AP + c] = f2bf(x[(size_t)sTok[m] * D + c]);
  }
  // visibility via the first __syncthreads() of the pipeline below

  // ---------------- GEMM1: h = relu(xA @ W1 + b1); N=1024 in 128-col chunks,
  // K=512 in 64-row double-buffered async stages (2 WMMA per stage)
  const float* Wp1 = W1 + (size_t)pair * D * H2;
  for (int chunk = 0; chunk < 8; ++chunk) {
    const float* Wcol = Wp1 + chunk * 128;
    v8f acc = {};
    stage_async(Wcol, H2, bBase0, w, l);                       // prologue: ks=0
    for (int ks = 0; ks < 8; ++ks) {
      if (ks + 1 < 8) {
        stage_async(Wcol + (size_t)(ks + 1) * 64 * H2, H2,
                    ((ks + 1) & 1) ? bBase1 : bBase0, w, l);
        asm volatile("s_wait_asynccnt 0x8" ::: "memory");      // prev stage done
      } else {
        asm volatile("s_wait_asynccnt 0x0" ::: "memory");
      }
      __syncthreads();
      const float* Bc = &Bfp[ks & 1][0];
#pragma unroll
      for (int sub = 0; sub < 2; ++sub) {
        FragBF a;
        const u16* ap = &Abf[ln * AP + ks * 64 + sub * 32 + hi * 8];
        a.u[0] = *(const uint4*)ap;
        a.u[1] = *(const uint4*)(ap + 16);
        v16bf bv;
        const float* bp = Bc + (sub * 32 + hi * 8) * BPITCH + (w * 16 + ln);
#pragma unroll
        for (int j = 0; j < 16; ++j) bv[j] = (__bf16)bp[(j + (j >> 3) * 8) * BPITCH];
        acc = __builtin_amdgcn_wmma_f32_16x16x32_bf16(
            false, a.v, false, bv, (short)0, acc, false, false);
      }
    }
    // epilogue: +b1, relu, bf16 -> Hs
    int nGlob = chunk * 128 + w * 16 + ln;
    float bb = b1[(size_t)pair * H2 + nGlob];
#pragma unroll
    for (int v = 0; v < 8; ++v) {
      float hv = acc[v] + bb;
      hv = hv > 0.f ? hv : 0.f;
      Hs[(v + hi * 8) * HP + nGlob] = f2bf(hv);
    }
  }

  // ---------------- GEMM2: y = h @ W2 + b2; N=512 in 128-col chunks,
  // K=1024 in 64-row double-buffered async stages
  const float* Wp2 = W2 + (size_t)pair * H2 * D;
  for (int c2 = 0; c2 < 4; ++c2) {
    const float* Wcol = Wp2 + c2 * 128;
    v8f acc = {};
    stage_async(Wcol, D, bBase0, w, l);
    for (int ks = 0; ks < 16; ++ks) {
      if (ks + 1 < 16) {
        stage_async(Wcol + (size_t)(ks + 1) * 64 * D, D,
                    ((ks + 1) & 1) ? bBase1 : bBase0, w, l);
        asm volatile("s_wait_asynccnt 0x8" ::: "memory");
      } else {
        asm volatile("s_wait_asynccnt 0x0" ::: "memory");
      }
      __syncthreads();
      const float* Bc = &Bfp[ks & 1][0];
#pragma unroll
      for (int sub = 0; sub < 2; ++sub) {
        FragBF a;
        const u16* ap = &Hs[ln * HP + ks * 64 + sub * 32 + hi * 8];
        a.u[0] = *(const uint4*)ap;
        a.u[1] = *(const uint4*)(ap + 16);
        v16bf bv;
        const float* bp = Bc + (sub * 32 + hi * 8) * BPITCH + (w * 16 + ln);
#pragma unroll
        for (int j = 0; j < 16; ++j) bv[j] = (__bf16)bp[(j + (j >> 3) * 8) * BPITCH];
        acc = __builtin_amdgcn_wmma_f32_16x16x32_bf16(
            false, a.v, false, bv, (short)0, acc, false, false);
      }
    }
    // epilogue: +b2, scale by routing weight, atomic accumulate
    int nGlob = c2 * 128 + w * 16 + ln;
    float bb = b2[(size_t)pair * D + nGlob];
#pragma unroll
    for (int v = 0; v < 8; ++v) {
      int m = v + hi * 8;
      float wgt = sWt[m];
      if (wgt != 0.f) {
        float val = (acc[v] + bb) * wgt;
        __hip_atomic_fetch_add(&out[(size_t)sTok[m] * D + nGlob], val,
                               __ATOMIC_RELAXED, __HIP_MEMORY_SCOPE_AGENT);
      }
    }
  }
}

// ---------------------------------------------------------------------------
extern "C" void kernel_launch(void* const* d_in, const int* in_sizes, int n_in,
                              void* d_out, int out_size, void* d_ws, size_t ws_size,
                              hipStream_t stream) {
  (void)n_in; (void)ws_size;
  const float* x  = (const float*)d_in[0];
  const float* g1 = (const float*)d_in[1];
  const float* g2 = (const float*)d_in[2];
  const float* W1 = (const float*)d_in[3];
  const float* b1 = (const float*)d_in[4];
  const float* W2 = (const float*)d_in[5];
  const float* b2 = (const float*)d_in[6];
  float* out = (float*)d_out;

  char* ws = (char*)d_ws;
  int*   cnt    = (int*)ws;
  int*   tokIdx = (int*)(ws + 256);
  float* tokW   = (float*)(ws + 256 + (size_t)NPAIR * MAXTOK * sizeof(int));

  const int T = in_sizes[0] / D;          // 1024 tokens
  const int tiles = (T + 15) / 16;        // 64

  moe_zero<<<(out_size + 255) / 256, 256, 0, stream>>>(out, out_size, cnt);
  moe_route<<<T, 64, 0, stream>>>(x, g1, g2, cnt, tokIdx, tokW);
  moe_expert<<<dim3(tiles, NPAIR), 256, 0, stream>>>(x, W1, b1, W2, b2,
                                                     cnt, tokIdx, tokW, out);
}
